// BodyOrderedModel_10943576670893
// MI455X (gfx1250) — matrix operand order repaired and checked by reading, not compiled, not run on hardware
//
#include <hip/hip_runtime.h>

// ---------------------------------------------------------------------------
// CDNA5 (gfx1250) body-ordered model fwd+bwd.
// Dead-code-eliminated to the l=0 radial chain (energy reads only lm=0, the
// message reads only lm=0, mixed/skip are diagonal in m => forces are purely
// radial). Channel-mixing GEMMs on v_wmma_f32_16x16x32_f16 with TDM-staged
// pre-packed f16 weight operands in LDS. The skip contraction is a single
// K=320 GEMM whose a-th K-chunk A-operand is attrs[n,a]*f[n,c] (per-lane
// scalar scaling of the A rows -> pure WMMA accumulation chain).
// ---------------------------------------------------------------------------

typedef __attribute__((ext_vector_type(16))) _Float16     v16h;
typedef __attribute__((ext_vector_type(8)))  float        v8f;
typedef __attribute__((ext_vector_type(4)))  unsigned int u32x4;
typedef __attribute__((ext_vector_type(4)))  int          i32x4;
typedef __attribute__((ext_vector_type(8)))  int          i32x8;

#define CCH   32
#define AATTR 10
#define NBES  8
#define RMAXF 5.0f
#define S2R   0.6324555320336759f   // sqrt(2/5)
#define PI_F  3.14159265358979f

#define NMAT_LAYER 11                              // Wmsg + 10 x Wskip[a]
#define MAT_HALFS  1024                            // one 32x32 B image (f16)
#define LAYER_HALFS (NMAT_LAYER * MAT_HALFS)       // 11264 halfs = 22528 B
#define PACK_HALFS ((2 * NMAT_LAYER + 1) * MAT_HALFS)  // + Wmsg0^T image

#if defined(__has_builtin)
#if __has_builtin(__builtin_amdgcn_tensor_load_to_lds)
#define HAVE_TDM 1
#endif
#endif
#ifndef HAVE_TDM
#define HAVE_TDM 0
#endif

static __device__ __forceinline__ v8f vzero8() {
  v8f z;
#pragma unroll
  for (int i = 0; i < 8; ++i) z[i] = 0.0f;
  return z;
}

static __device__ __forceinline__ v8f wmma16(v16h a, v16h b, v8f c) {
  return __builtin_amdgcn_wmma_f32_16x16x32_f16(false, a, false, b, (short)0, c,
                                                false, false);
}

#if HAVE_TDM
// One 1-D TDM descriptor: copy `halfs` f16 elements global -> LDS.
// D# per ISA 8.3/8.4: group0 {count=1, lds_addr, global_addr, type=2},
// group1 {data_size=1(2B), tensor_dim0=tile_dim0=halfs, dims1=1, stride0=halfs}.
static __device__ __forceinline__ void tdm_load_to_lds(const void* gsrc,
                                                       void* lds_dst, int halfs) {
  unsigned long long ga = (unsigned long long)(size_t)gsrc;
  unsigned lds = (unsigned)(size_t)lds_dst;   // low 32 bits = LDS byte address
  u32x4 g0;
  g0[0] = 1u;                                   // count = 1 (valid)
  g0[1] = lds;                                  // lds_addr [63:32]
  g0[2] = (unsigned)ga;                         // global_addr [95:64]
  g0[3] = (unsigned)((ga >> 32) & 0x01FFFFFFull) | (2u << 30);  // [120:96] | type=2
  i32x8 g1;
  g1[0] = (int)(1u << 16);                      // data_size = 1 (2 bytes)
  g1[1] = (int)(((unsigned)halfs & 0xFFFFu) << 16);   // tensor_dim0[15:0] @ [63:48]
  g1[2] = (int)((((unsigned)halfs >> 16) & 0xFFFFu) | (1u << 16)); // td0 hi | td1=1
  g1[3] = (int)(((unsigned)halfs & 0xFFFFu) << 16);   // tile_dim0 @ [127:112]
  g1[4] = 1;                                    // tile_dim1 = 1
  g1[5] = halfs;                                // tensor_dim0_stride[31:0]
  g1[6] = 0;
  g1[7] = 0;
  i32x4 gz;
  gz[0] = 0; gz[1] = 0; gz[2] = 0; gz[3] = 0;
#if __clang_major__ >= 23
  i32x8 gz8;
#pragma unroll
  for (int i = 0; i < 8; ++i) gz8[i] = 0;
  __builtin_amdgcn_tensor_load_to_lds(g0, g1, gz, gz, gz8, 0);
#else
  __builtin_amdgcn_tensor_load_to_lds(g0, g1, gz, gz, 0);
#endif
  __builtin_amdgcn_s_wait_tensorcnt(0);
}
#endif

// Load the 16 f32 values of this lane's A-row (ISA 16-bit A 16x32 layout:
// lane L(<16) row L holds K {8h..8h+7} U {16+8h..23+8h}, h=lane/16).
static __device__ __forceinline__ void loadArow(const float* __restrict__ src,
                                                int gr, int h, float* rowv) {
  const float4* p = (const float4*)(src + (size_t)gr * CCH);
  float4 f0 = p[(h << 1) + 0];
  float4 f1 = p[(h << 1) + 1];
  float4 f2 = p[(h << 1) + 4];
  float4 f3 = p[(h << 1) + 5];
  rowv[0] = f0.x;  rowv[1] = f0.y;  rowv[2] = f0.z;  rowv[3] = f0.w;
  rowv[4] = f1.x;  rowv[5] = f1.y;  rowv[6] = f1.z;  rowv[7] = f1.w;
  rowv[8] = f2.x;  rowv[9] = f2.y;  rowv[10] = f2.z; rowv[11] = f2.w;
  rowv[12] = f3.x; rowv[13] = f3.y; rowv[14] = f3.z; rowv[15] = f3.w;
}

static __device__ __forceinline__ v16h cvtA(const float* rowv, float scale) {
  v16h a;
#pragma unroll
  for (int t = 0; t < 16; ++t) a[t] = (_Float16)(scale * rowv[t]);
  return a;
}

// ---------------------------------------------------------------------------
__global__ void zero_kernel(float* __restrict__ p, long n) {
  long i = (long)blockIdx.x * blockDim.x + threadIdx.x;
  long stride = (long)gridDim.x * blockDim.x;
  for (; i < n; i += stride) p[i] = 0.0f;
}

// Pre-pack WMMA-ready f16 B-operand images: [dhalf(2)][lane(32)][t(16)].
// mats: 0=Wmsg0, 1..10=Wskip0[a], 11=Wmsg1, 12..21=Wskip1[a], 22=Wmsg0^T.
__global__ void pack_weights_kernel(const float* __restrict__ Wmsg0,
                                    const float* __restrict__ Wskip0,
                                    const float* __restrict__ Wmsg1,
                                    const float* __restrict__ Wskip1,
                                    _Float16* __restrict__ pack) {
  int idx = blockIdx.x * blockDim.x + threadIdx.x;
  if (idx >= PACK_HALFS) return;
  int mat = idx >> 10;
  int r = idx & 1023;
  int hh = r >> 9;              // d-half
  int lane = (r >> 4) & 31;
  int t = r & 15;
  int K = ((lane >> 4) << 3) + (t & 7) + ((t >> 3) << 4);  // B row (contraction index)
  int Nn = (lane & 15) + (hh << 4);        // B column
  float v;
  if (mat == 22)      v = Wmsg0[Nn * CCH + K];                       // transposed
  else if (mat == 0)  v = Wmsg0[K * CCH + Nn];
  else if (mat <= 10) v = Wskip0[(mat - 1) * 1024 + K * CCH + Nn];
  else if (mat == 11) v = Wmsg1[K * CCH + Nn];
  else                v = Wskip1[(mat - 12) * 1024 + K * CCH + Nn];
  pack[idx] = (_Float16)v;
}

// per-edge: r -> bessel*cutoff features (E x 8)
__global__ void edge_geom_kernel(const float* __restrict__ pos,
                                 const float* __restrict__ shifts,
                                 const int* __restrict__ ei,
                                 float* __restrict__ ef, int E) {
  int e = blockIdx.x * blockDim.x + threadIdx.x;
  if (e >= E) return;
  int s = ei[e], r_ = ei[E + e];
  float vx = pos[r_ * 3 + 0] - pos[s * 3 + 0] + shifts[e * 3 + 0];
  float vy = pos[r_ * 3 + 1] - pos[s * 3 + 1] + shifts[e * 3 + 1];
  float vz = pos[r_ * 3 + 2] - pos[s * 3 + 2] + shifts[e * 3 + 2];
  float r = sqrtf(vx * vx + vy * vy + vz * vz);
  float sr = fmaxf(r, 1e-9f);
  float x = r * (1.0f / RMAXF);
  float cut = 0.0f;
  if (x < 1.0f) {
    float x2 = x * x, x4 = x2 * x2, x6 = x4 * x2, x7 = x6 * x, x8 = x6 * x2;
    cut = 1.0f - 28.0f * x6 + 48.0f * x7 - 21.0f * x8;   // P=6
  }
  float inv = 1.0f / sr;
#pragma unroll
  for (int k = 0; k < NBES; ++k) {
    float w = (float)(k + 1) * (PI_F / RMAXF);
    ef[(size_t)e * NBES + k] = S2R * sinf(w * sr) * inv * cut;
  }
}

// s0 = attrs @ W_emb ; E0 = attrs @ atomic_energies -> per-graph energy
__global__ void node_init_kernel(const float* __restrict__ attrs,
                                 const float* __restrict__ Wemb,
                                 const float* __restrict__ ae,
                                 const int* __restrict__ batch,
                                 float* __restrict__ s0,
                                 float* __restrict__ energy, int N) {
  int idx = blockIdx.x * blockDim.x + threadIdx.x;
  if (idx >= N * CCH) return;
  int n = idx >> 5, c = idx & 31;
  float v = 0.0f;
#pragma unroll
  for (int a = 0; a < AATTR; ++a) v += attrs[n * AATTR + a] * Wemb[a * CCH + c];
  s0[idx] = v;
  if (c == 0) {
    float e0 = 0.0f;
#pragma unroll
    for (int a = 0; a < AATTR; ++a) e0 += attrs[n * AATTR + a] * ae[a];
    atomicAdd(&energy[batch[n]], e0);
  }
}

// per-edge message (l=0 only): agg[recv,c] += s_prev[send,c]*Rw[c]
// (layer 2 also) T[send,c] += Rw[c] for the backward pass.
__global__ void edge_message_kernel(const float* __restrict__ ef,
                                    const int* __restrict__ ei,
                                    const float* __restrict__ sprev,
                                    const float* __restrict__ Wrad,
                                    float* __restrict__ agg,
                                    float* __restrict__ Tsend, int E) {
  __shared__ float Wcol[NBES][CCH];
  int tid = threadIdx.x;
  {
    int k = tid >> 5, c = tid & 31;            // 256 threads == 8*32
    Wcol[k][c] = Wrad[k * 96 + 3 * c];         // column l=0 of (8,96)
  }
  __syncthreads();
  int lane = tid & 31, wave = tid >> 5;
  int e = blockIdx.x * 8 + wave;
  if (e >= E) return;
  __builtin_prefetch(ef + (size_t)e * NBES, 0, 0);   // global_prefetch_b8
  int send = ei[e], recv = ei[E + e];
  float rw = 0.0f;
#pragma unroll
  for (int k = 0; k < NBES; ++k) rw += ef[(size_t)e * NBES + k] * Wcol[k][lane];
  float val = sprev[(size_t)send * CCH + lane] * rw;
  atomicAdd(&agg[(size_t)recv * CCH + lane], val);
  if (Tsend) atomicAdd(&Tsend[(size_t)send * CCH + lane], rw);
}

// feats = agg @ Wmsg + sum_a (attrs[:,a]*fprev) @ Wskip[a]
// One chained WMMA accumulator pair per d-half; B operands TDM-staged in LDS.
__global__ void node_update_kernel(const float* __restrict__ agg,
                                   const float* __restrict__ fprev,
                                   const float* __restrict__ attrs,
                                   const _Float16* __restrict__ packL,
                                   float* __restrict__ fout, int N) {
  __shared__ v16h smemB[NMAT_LAYER * 64];      // 11 matrices x 2 halves x 32 lanes
#if HAVE_TDM
  if ((threadIdx.x >> 5) == 0)
    tdm_load_to_lds(packL, (void*)smemB, LAYER_HALFS);
#else
  for (int i = threadIdx.x; i < LAYER_HALFS / 2; i += blockDim.x)
    ((unsigned*)smemB)[i] = ((const unsigned*)packL)[i];
#endif
  __syncthreads();

  int lane = threadIdx.x & 31;
  int wave = threadIdx.x >> 5;
  int tile = blockIdx.x * (blockDim.x >> 5) + wave;
  int ntiles = (N + 15) >> 4;
  if (tile < ntiles) {                         // wave-uniform: EXEC stays full
    int row0 = tile << 4;
    int h = lane >> 4;
    int row = lane & 15;
    int gr = row0 + row;
    if (gr > N - 1) gr = N - 1;                // clamp; clamped rows never stored

    // this lane's A-row of agg (for mixed) and fprev (for skip), in f32
    float rowA[16], rowF[16];
    loadArow(agg, gr, h, rowA);
    loadArow(fprev, gr, h, rowF);

    // this lane's own attrs row (5 aligned float2 loads)
    float attr[AATTR];
    const float2* pa = (const float2*)(attrs + (size_t)gr * AATTR);
#pragma unroll
    for (int j = 0; j < 5; ++j) {
      float2 t2 = pa[j];
      attr[2 * j] = t2.x;
      attr[2 * j + 1] = t2.y;
    }

    v16h aA = cvtA(rowA, 1.0f);
    v8f acc0 = wmma16(aA, smemB[lane], vzero8());       // mixed, d-half 0
    v8f acc1 = wmma16(aA, smemB[32 + lane], vzero8());  // mixed, d-half 1

#pragma unroll
    for (int a = 0; a < AATTR; ++a) {
      v16h As = cvtA(rowF, attr[a]);                    // attrs[n,a]*f[n,:]
      int mb = (1 + a) * 64;
      acc0 = wmma16(As, smemB[mb + lane], acc0);        // chained accumulation
      acc1 = wmma16(As, smemB[mb + 32 + lane], acc1);
    }

    int nlo = lane & 15;
#pragma unroll
    for (int i = 0; i < 8; ++i) {
      int m = row0 + i + (h << 3);
      if (m < N) {
        fout[(size_t)m * CCH + nlo]      = acc0[i];
        fout[(size_t)m * CCH + 16 + nlo] = acc1[i];
      }
    }
  }
}

// per-node readout of both layers into per-graph energies
__global__ void readout_kernel(const float* __restrict__ f1,
                               const float* __restrict__ f2,
                               const float* __restrict__ Wr0,
                               const float* __restrict__ Wr1,
                               const int* __restrict__ batch,
                               float* __restrict__ energy, int N) {
  int n = blockIdx.x * blockDim.x + threadIdx.x;
  if (n >= N) return;
  float e = 0.0f;
#pragma unroll
  for (int d = 0; d < CCH; ++d)
    e += f1[(size_t)n * CCH + d] * Wr0[d] + f2[(size_t)n * CCH + d] * Wr1[d];
  atomicAdd(&energy[batch[n]], e);
}

// backward constants: ga2[c] = sum_d Wr1[d]*Wmsg1[c,d];
//                     V[a,c] = sum_d Wr1[d]*Wskip1[a,c,d]
__global__ void bwd_prep_kernel(const float* __restrict__ Wr1,
                                const float* __restrict__ Wmsg1,
                                const float* __restrict__ Wskip1,
                                float* __restrict__ ga2,
                                float* __restrict__ V) {
  int t = threadIdx.x;
  if (t < 32) {
    float s = 0.0f;
#pragma unroll
    for (int d = 0; d < CCH; ++d) s += Wr1[d] * Wmsg1[t * CCH + d];
    ga2[t] = s;
  } else if (t < 32 + AATTR * CCH) {
    int i = t - 32, a = i >> 5, c = i & 31;
    float s = 0.0f;
#pragma unroll
    for (int d = 0; d < CCH; ++d) s += Wr1[d] * Wskip1[a * 1024 + c * CCH + d];
    V[i] = s;
  }
}

// g1[n,c] = Wr0[c] + sum_a attrs[n,a]*V[a,c] + ga2[c]*T[n,c]
__global__ void g1_kernel(const float* __restrict__ attrs,
                          const float* __restrict__ Wr0,
                          const float* __restrict__ V,
                          const float* __restrict__ ga2,
                          const float* __restrict__ T,
                          float* __restrict__ g1, int N) {
  int idx = blockIdx.x * blockDim.x + threadIdx.x;
  if (idx >= N * CCH) return;
  int n = idx >> 5, c = idx & 31;
  float v = Wr0[c] + ga2[c] * T[idx];
#pragma unroll
  for (int a = 0; a < AATTR; ++a) v += attrs[n * AATTR + a] * V[a * CCH + c];
  g1[idx] = v;
}

// ga1 = g1 @ Wmsg0^T  (WMMA; pre-packed transposed B image)
__global__ void ga1_kernel(const float* __restrict__ g1,
                           const _Float16* __restrict__ packT,
                           float* __restrict__ ga1, int N) {
  int lane = threadIdx.x & 31;
  int wave = threadIdx.x >> 5;
  int tile = blockIdx.x * (blockDim.x >> 5) + wave;
  int ntiles = (N + 15) >> 4;
  if (tile >= ntiles) return;
  int row0 = tile << 4;
  int h = lane >> 4;
  int row = lane & 15;
  int gr = row0 + row;
  if (gr > N - 1) gr = N - 1;
  float rowG[16];
  loadArow(g1, gr, h, rowG);
  v16h aG = cvtA(rowG, 1.0f);
  const v16h* BT = (const v16h*)packT;
  v8f c0 = wmma16(aG, BT[lane], vzero8());
  v8f c1 = wmma16(aG, BT[32 + lane], vzero8());
  int nlo = lane & 15;
#pragma unroll
  for (int i = 0; i < 8; ++i) {
    int m = row0 + i + (h << 3);
    if (m < N) {
      ga1[(size_t)m * CCH + nlo]      = c0[i];
      ga1[(size_t)m * CCH + 16 + nlo] = c1[i];
    }
  }
}

// per-edge force: dE/def_k = sum_c (ga1[recv]s0[send] + ga2*f1[send])*Wrad[k,3c]
// chained through analytic d(bessel*cut)/dr; +-dEdr*unit atomics into forces.
__global__ void edge_force_kernel(const float* __restrict__ pos,
                                  const float* __restrict__ shifts,
                                  const int* __restrict__ ei,
                                  const float* __restrict__ s0,
                                  const float* __restrict__ f1,
                                  const float* __restrict__ ga1,
                                  const float* __restrict__ ga2,
                                  const float* __restrict__ Wrad0,
                                  const float* __restrict__ Wrad1,
                                  float* __restrict__ force, int E) {
  __shared__ float W0[NBES][CCH];
  __shared__ float W1[NBES][CCH];
  int tid = threadIdx.x;
  {
    int k = tid >> 5, c = tid & 31;
    W0[k][c] = Wrad0[k * 96 + 3 * c];
    W1[k][c] = Wrad1[k * 96 + 3 * c];
  }
  __syncthreads();
  int lane = tid & 31, wave = tid >> 5;
  int e = blockIdx.x * 8 + wave;
  if (e >= E) return;
  int sn = ei[e], rn = ei[E + e];
  __builtin_prefetch(pos + (size_t)sn * 3, 0, 0);
  float alpha = ga1[(size_t)rn * CCH + lane] * s0[(size_t)sn * CCH + lane];
  float beta  = ga2[lane] * f1[(size_t)sn * CCH + lane];
  float p[NBES];
#pragma unroll
  for (int k = 0; k < NBES; ++k) p[k] = alpha * W0[k][lane] + beta * W1[k][lane];
#pragma unroll
  for (int off = 16; off > 0; off >>= 1) {
#pragma unroll
    for (int k = 0; k < NBES; ++k) p[k] += __shfl_xor(p[k], off, 32);
  }
  float vx = pos[rn * 3 + 0] - pos[sn * 3 + 0] + shifts[e * 3 + 0];
  float vy = pos[rn * 3 + 1] - pos[sn * 3 + 1] + shifts[e * 3 + 1];
  float vz = pos[rn * 3 + 2] - pos[sn * 3 + 2] + shifts[e * 3 + 2];
  float r = sqrtf(vx * vx + vy * vy + vz * vz);
  float sr = fmaxf(r, 1e-9f);
  float x = r * (1.0f / RMAXF);
  float cut = 0.0f, dcut = 0.0f;
  if (x < 1.0f) {
    float x2 = x * x, x4 = x2 * x2, x5 = x4 * x, x6 = x4 * x2, x7 = x6 * x, x8 = x6 * x2;
    cut  = 1.0f - 28.0f * x6 + 48.0f * x7 - 21.0f * x8;
    dcut = (-168.0f * x5 + 336.0f * x6 - 168.0f * x7) * (1.0f / RMAXF);
  }
  float inv = 1.0f / sr;
  float dEdr = 0.0f;
#pragma unroll
  for (int k = 0; k < NBES; ++k) {
    float w = (float)(k + 1) * (PI_F / RMAXF);
    float sw = sinf(w * sr), cw = cosf(w * sr);
    float B  = S2R * sw * inv;
    float dB = S2R * (w * cw * inv - sw * inv * inv);
    dEdr += p[k] * (dB * cut + B * dcut);
  }
  float ux = vx * inv, uy = vy * inv, uz = vz * inv;
  if (lane < 3) {
    float u = (lane == 0) ? ux : ((lane == 1) ? uy : uz);
    atomicAdd(&force[(size_t)rn * 3 + lane], -dEdr * u);
    atomicAdd(&force[(size_t)sn * 3 + lane],  dEdr * u);
  }
}

// ---------------------------------------------------------------------------
extern "C" void kernel_launch(void* const* d_in, const int* in_sizes, int n_in,
                              void* d_out, int out_size, void* d_ws, size_t ws_size,
                              hipStream_t stream) {
  (void)n_in; (void)ws_size;
  const float* pos    = (const float*)d_in[0];
  const float* attrs  = (const float*)d_in[1];
  const float* shifts = (const float*)d_in[2];
  const float* Wemb   = (const float*)d_in[3];
  const float* ae     = (const float*)d_in[4];
  const float* Wrad   = (const float*)d_in[5];   // (2,8,96)
  const float* Wmsg   = (const float*)d_in[6];   // (2,3,32,32)
  const float* Wskip  = (const float*)d_in[7];   // (2,3,10,32,32)
  const float* Wro    = (const float*)d_in[8];   // (2,32)
  const int*   ei     = (const int*)d_in[9];     // (2,E)
  const int*   batch  = (const int*)d_in[10];

  const int N = in_sizes[0] / 3;
  const int E = in_sizes[9] / 2;

  float* ws = (float*)d_ws;
  size_t off = 0;
  float* ef   = ws + off; off += (size_t)E * NBES;
  float* s0   = ws + off; off += (size_t)N * CCH;
  float* f1   = ws + off; off += (size_t)N * CCH;
  float* f2   = ws + off; off += (size_t)N * CCH;
  float* agg1 = ws + off; off += (size_t)N * CCH;   // agg1,agg2,T contiguous
  float* agg2 = ws + off; off += (size_t)N * CCH;
  float* Tsum = ws + off; off += (size_t)N * CCH;
  float* g1   = ws + off; off += (size_t)N * CCH;
  float* ga1  = ws + off; off += (size_t)N * CCH;
  float* ga2  = ws + off; off += CCH;
  float* Vac  = ws + off; off += AATTR * CCH;
  off = (off + 7) & ~(size_t)7;                     // 32B-align the f16 pack
  _Float16* pack = (_Float16*)(ws + off); off += PACK_HALFS / 2;

  float* energy = (float*)d_out;
  float* force  = (float*)d_out + (out_size - 3 * N);

  const float* Wrad0  = Wrad;
  const float* Wrad1  = Wrad + NBES * 96;
  const float* Wmsg0  = Wmsg;                       // [t=0][l=0]
  const float* Wmsg1  = Wmsg + 3 * CCH * CCH;       // [t=1][l=0]
  const float* Wskip0 = Wskip;                      // [t=0][l=0]
  const float* Wskip1 = Wskip + 3 * AATTR * CCH * CCH;
  const float* Wro0   = Wro;
  const float* Wro1   = Wro + CCH;

  const int ntiles = (N + 15) / 16;
  dim3 tileGrid((ntiles + 3) / 4);                  // 4 waves / block

  zero_kernel<<<dim3((out_size + 255) / 256), 256, 0, stream>>>(energy, (long)out_size);
  zero_kernel<<<dim3(((long)N * CCH * 3 + 255) / 256), 256, 0, stream>>>(agg1, (long)N * CCH * 3);

  pack_weights_kernel<<<dim3((PACK_HALFS + 255) / 256), 256, 0, stream>>>(
      Wmsg0, Wskip0, Wmsg1, Wskip1, pack);
  edge_geom_kernel<<<dim3((E + 255) / 256), 256, 0, stream>>>(pos, shifts, ei, ef, E);
  node_init_kernel<<<dim3((N * CCH + 255) / 256), 256, 0, stream>>>(attrs, Wemb, ae, batch, s0, energy, N);

  // layer 1
  edge_message_kernel<<<dim3((E + 7) / 8), 256, 0, stream>>>(ef, ei, s0, Wrad0, agg1, (float*)nullptr, E);
  node_update_kernel<<<tileGrid, 128, 0, stream>>>(agg1, s0, attrs, pack, f1, N);

  // layer 2 (also accumulates T for backward)
  edge_message_kernel<<<dim3((E + 7) / 8), 256, 0, stream>>>(ef, ei, f1, Wrad1, agg2, Tsum, E);
  node_update_kernel<<<tileGrid, 128, 0, stream>>>(agg2, f1, attrs, pack + LAYER_HALFS, f2, N);

  readout_kernel<<<dim3((N + 255) / 256), 256, 0, stream>>>(f1, f2, Wro0, Wro1, batch, energy, N);

  // backward (forces): radial-only analytic chain
  bwd_prep_kernel<<<dim3(1), 384, 0, stream>>>(Wro1, Wmsg1, Wskip1, ga2, Vac);
  g1_kernel<<<dim3((N * CCH + 255) / 256), 256, 0, stream>>>(attrs, Wro0, Vac, ga2, Tsum, g1, N);
  ga1_kernel<<<tileGrid, 128, 0, stream>>>(g1, pack + 22 * MAT_HALFS, ga1, N);
  edge_force_kernel<<<dim3((E + 7) / 8), 256, 0, stream>>>(pos, shifts, ei, s0, f1, ga1, ga2,
                                                           Wrad0, Wrad1, force, E);
}
